// Decoder_54786602828308
// MI455X (gfx1250) — matrix operand order repaired and checked by reading, not compiled, not run on hardware
//
#include <hip/hip_runtime.h>

// ---------------------------------------------------------------------------
// Pointer-network decoder for MI455X (gfx1250, wave32, WMMA).
//
// Decomposition: batch rows are independent across the 128-step scan, so the
// whole recurrence runs in 16 persistent workgroups (16 batch rows each, the
// WMMA M-tile), with only workgroup barriers between stages. Loop-invariant
// work (x-gates GEMM, ctx = W_ctx*context + b_ctx) is precomputed by separate
// kernels. GEMMs use v_wmma_f32_16x16x32_bf16 (bf16 in, fp32 accumulate);
// recurrent state h/c, softmax, argmax stay fp32.
// ---------------------------------------------------------------------------

#define B_   256
#define L_   128
#define E_   256
#define H_   512
#define G4   2048   // 4*H

typedef __bf16 bf16_t;
typedef bf16_t bf16x16 __attribute__((ext_vector_type(16)));
typedef float  f32x8   __attribute__((ext_vector_type(8)));

union Frag {
  bf16x16 v;
  uint4   q[2];
};

// ---- fast transcendentals --------------------------------------------------
__device__ __forceinline__ float fast_tanh(float x) {
#if __has_builtin(__builtin_amdgcn_tanhf)
  return __builtin_amdgcn_tanhf(x);
#elif __has_builtin(__builtin_amdgcn_tanh_f32)
  return __builtin_amdgcn_tanh_f32(x);
#else
  float e = __expf(2.0f * x);                       // v_exp_f32
  return 1.0f - 2.0f * __builtin_amdgcn_rcpf(e + 1.0f);
#endif
}

__device__ __forceinline__ float sigm(float x) {
  return __builtin_amdgcn_rcpf(1.0f + __expf(-x));
}

// ---- WMMA fragment loaders (layouts per CDNA5 ISA §7.12.2) ------
// A: 16x32 bf16 tile, row-major [16][512] in LDS.
//   lane L: M = L&15 ; K-halves: lanes 0-15 -> K {0..7,16..23},
//                                lanes16-31 -> K {8..15,24..31}
__device__ __forceinline__ void load_afrag(const bf16_t* __restrict__ base,
                                           int kb, int lane, Frag& f) {
  const bf16_t* row = base + (lane & 15) * H_;
  int k0 = kb + ((lane >> 4) & 1) * 8;
  f.q[0] = *(const uint4*)(row + k0);
  f.q[1] = *(const uint4*)(row + k0 + 16);
}

// B: 32x16 bf16 tile from global weights stored [N][K=512] row-major.
//   lane L: N = n0 + (L&15) ; lanes 0-15 hold K {0..7,16..23}, lanes16-31 {8..15,24..31}
__device__ __forceinline__ void load_bfrag(const bf16_t* __restrict__ W,
                                           int n0, int kb, int lane, Frag& f) {
  int n  = n0 + (lane & 15);
  const bf16_t* p = W + (size_t)n * H_ + kb + ((lane >> 4) & 1) * 8;
  f.q[0] = *(const uint4*)(p);
  f.q[1] = *(const uint4*)(p + 16);
}

// full K=512 sweep for one 16x16 output tile
__device__ __forceinline__ f32x8 wmma_row(const bf16_t* __restrict__ A_lds,
                                          const bf16_t* __restrict__ W,
                                          int n0, int lane) {
  f32x8 acc = {0.f, 0.f, 0.f, 0.f, 0.f, 0.f, 0.f, 0.f};
#pragma unroll
  for (int kb = 0; kb < H_; kb += 32) {
    Frag a, b;
    load_afrag(A_lds, kb, lane, a);
    load_bfrag(W, n0, kb, lane, b);
    acc = __builtin_amdgcn_wmma_f32_16x16x32_bf16(
        false, a.v, false, b.v, (short)0, acc, false, false);
  }
  return acc;
}

// ---------------------------------------------------------------------------
// Kernel 1: convert weights to bf16 (and fold W_out[:, :H] + W_out[:, H:]).
// ---------------------------------------------------------------------------
__global__ void conv_kernel(const float* __restrict__ Whh_f,
                            const float* __restrict__ Winp_f,
                            const float* __restrict__ Wctx_f,
                            const float* __restrict__ Wout_f,
                            bf16_t* __restrict__ Whh, bf16_t* __restrict__ Winp,
                            bf16_t* __restrict__ Wctx, bf16_t* __restrict__ Wo) {
  int gid = blockIdx.x * 256 + threadIdx.x;
  if (gid < G4 * H_) { Whh[gid] = (bf16_t)Whh_f[gid]; return; }
  int i = gid - G4 * H_;
  if (i < H_ * H_) { Winp[i] = (bf16_t)Winp_f[i]; return; }
  i -= H_ * H_;
  if (i < H_ * H_) { Wctx[i] = (bf16_t)Wctx_f[i]; return; }
  i -= H_ * H_;
  int n = i >> 9, k = i & 511;
  Wo[i] = (bf16_t)(Wout_f[n * 1024 + k] + Wout_f[n * 1024 + 512 + k]);
}

// ---------------------------------------------------------------------------
// Kernel 2: gx[b][n] = dec_inp[b] . W_ih[n] + b_ih[n] + b_hh[n]  (one-time)
// ---------------------------------------------------------------------------
__global__ void gx_kernel(const float* __restrict__ dec_inp,
                          const float* __restrict__ Wih,
                          const float* __restrict__ bih,
                          const float* __restrict__ bhh,
                          float* __restrict__ gx) {
  int gid = blockIdx.x * 256 + threadIdx.x;
  int b = gid >> 11, n = gid & 2047;
  const float* x = dec_inp + b * E_;
  const float* w = Wih + (size_t)n * E_;
  float acc = bih[n] + bhh[n];
#pragma unroll 4
  for (int e = 0; e < E_; ++e) acc += x[e] * w[e];
  gx[gid] = acc;
}

// ---------------------------------------------------------------------------
// Kernel 3: ctx[b][o][l] = sum_h W_ctx[o][h]*context[b][l][h] + b_ctx[o]
// WMMA: M-rows = (b,l) pairs, N = o, K = h.   grid = 2048 blocks.
// ---------------------------------------------------------------------------
__global__ __launch_bounds__(256)
void ctx_kernel(const float* __restrict__ context,
                const bf16_t* __restrict__ Wctx,
                const float* __restrict__ b_ctx,
                float* __restrict__ ctx_out) {
  __shared__ alignas(16) bf16_t Abf[16 * H_];
  int tid = threadIdx.x, lane = tid & 31, wave = tid >> 5;
  int R0 = blockIdx.x * 16;
#pragma unroll
  for (int k = 0; k < 32; ++k) {
    int idx = tid + 256 * k;
    int m = idx >> 9, h = idx & 511;
    int R = R0 + m, b = R >> 7, l = R & 127;
    Abf[m * H_ + h] = (bf16_t)context[((size_t)b * L_ + l) * H_ + h];
  }
  __syncthreads();
#pragma unroll 1
  for (int tt = 0; tt < 4; ++tt) {
    int n0 = wave * 64 + tt * 16;
    f32x8 acc = wmma_row(Abf, Wctx, n0, lane);
    int n = n0 + (lane & 15);
    int mb = (lane >> 4) * 8;
    float bc = b_ctx[n];
#pragma unroll
    for (int r = 0; r < 8; ++r) {
      int R = R0 + mb + r, b = R >> 7, l = R & 127;
      ctx_out[((size_t)b * H_ + n) * L_ + l] = acc[r] + bc;
    }
  }
}

// ---------------------------------------------------------------------------
// Kernel 4: the 128-step scan. 16 blocks x 256 threads (8 waves).
// Block owns batch rows [16*blk, 16*blk+16).  c-state lives in registers.
// ---------------------------------------------------------------------------
__global__ __launch_bounds__(256)
void scan_kernel(const float* __restrict__ h0, const float* __restrict__ c0,
                 const float* __restrict__ ctx, const float* __restrict__ gx,
                 const bf16_t* __restrict__ Whh, const bf16_t* __restrict__ Winp,
                 const bf16_t* __restrict__ Wo, const float* __restrict__ b_inp,
                 const float* __restrict__ b_out, const float* __restrict__ Vg,
                 float* __restrict__ outA, float* __restrict__ outP,
                 float* __restrict__ outHf, float* __restrict__ outCf) {
  // ---- LDS (≈154 KB, < 160 KB) ----
  __shared__ alignas(16) float  gbuf[16 * 1024];   // 64 KB gate half-buffer
  __shared__ alignas(16) bf16_t actA[16 * H_];     // 16 KB: h carry (bf16)
  __shared__ alignas(16) bf16_t actB[16 * H_];     // 16 KB: h_rnn / hs (bf16)
  __shared__ alignas(16) float  hp[16 * H_];       // 32 KB
  __shared__ float s_l[16 * L_];                   //  8 KB scores
  __shared__ float alpha_l[16 * L_];               //  8 KB
  __shared__ float mask_l[16 * L_];                //  8 KB
  __shared__ float V_l[H_];                        //  2 KB

  const int tid  = threadIdx.x;
  const int lane = tid & 31;
  const int wave = tid >> 5;
  const int b0   = blockIdx.x * 16;
  const float NEGBIG = -1.0e30f;

  float c_reg[32];
  // ---- init ----
#pragma unroll
  for (int k = 0; k < 32; ++k) {
    int idx = tid + 256 * k;
    int b = idx >> 9, j = idx & 511;
    actA[b * H_ + j] = (bf16_t)h0[(size_t)(b0 + b) * H_ + j];
    c_reg[k] = c0[(size_t)(b0 + b) * H_ + j];
  }
#pragma unroll
  for (int k = 0; k < 8; ++k) mask_l[tid + 256 * k] = 1.0f;
  V_l[tid] = Vg[tid];
  V_l[tid + 256] = Vg[tid + 256];
  __syncthreads();

#pragma unroll 1
  for (int t = 0; t < L_; ++t) {
    // ---- GEMM1 pass A: gate cols 0..1023 (i, f) ----
#pragma unroll 1
    for (int tt = 0; tt < 8; ++tt) {
      int n0 = wave * 128 + tt * 16;
      f32x8 acc = wmma_row(actA, Whh, n0, lane);
      int n = n0 + (lane & 15);
      int mb = (lane >> 4) * 8;
#pragma unroll
      for (int r = 0; r < 8; ++r) {
        int m = mb + r;
        gbuf[m * 1024 + n] = acc[r] + gx[(size_t)(b0 + m) * G4 + n];
      }
    }
    __syncthreads();

    // pointwise part 1: stash sigmoid(i), sigmoid(f)*c in registers
    float si[32], cpart[32];
#pragma unroll
    for (int k = 0; k < 32; ++k) {
      int idx = tid + 256 * k;
      int b = idx >> 9, j = idx & 511;
      si[k]    = sigm(gbuf[b * 1024 + j]);
      cpart[k] = sigm(gbuf[b * 1024 + 512 + j]) * c_reg[k];
    }
    __syncthreads();

    // ---- GEMM1 pass B: gate cols 1024..2047 (g, o) ----
#pragma unroll 1
    for (int tt = 0; tt < 8; ++tt) {
      int n0 = 1024 + wave * 128 + tt * 16;
      f32x8 acc = wmma_row(actA, Whh, n0, lane);
      int n = n0 + (lane & 15);
      int mb = (lane >> 4) * 8;
#pragma unroll
      for (int r = 0; r < 8; ++r) {
        int m = mb + r;
        gbuf[m * 1024 + (n - 1024)] = acc[r] + gx[(size_t)(b0 + m) * G4 + n];
      }
    }
    __syncthreads();

    // pointwise part 2: c_t, h_rnn
#pragma unroll
    for (int k = 0; k < 32; ++k) {
      int idx = tid + 256 * k;
      int b = idx >> 9, j = idx & 511;
      float cn = cpart[k] + si[k] * fast_tanh(gbuf[b * 1024 + j]);
      float hr = sigm(gbuf[b * 1024 + 512 + j]) * fast_tanh(cn);
      c_reg[k] = cn;
      actB[b * H_ + j] = (bf16_t)hr;
      if (t == L_ - 1) outCf[(size_t)(b0 + b) * H_ + j] = cn;
    }
    __syncthreads();

    // ---- GEMM2: hp = h_rnn @ W_inp.T + b_inp ----
#pragma unroll 1
    for (int tt = 0; tt < 4; ++tt) {
      int n0 = wave * 64 + tt * 16;
      f32x8 acc = wmma_row(actB, Winp, n0, lane);
      int n = n0 + (lane & 15);
      int mb = (lane >> 4) * 8;
      float bi = b_inp[n];
#pragma unroll
      for (int r = 0; r < 8; ++r) hp[(mb + r) * H_ + n] = acc[r] + bi;
    }
    __syncthreads();

    // ---- attention scores: s[b][l] = sum_h V[h]*tanh(hp[b][h]+ctx[b][h][l])
#pragma unroll 1
    for (int k = 0; k < 8; ++k) {
      int idx = tid + 256 * k;
      int b = idx >> 7, l = idx & 127;
      const float* cp  = ctx + (size_t)(b0 + b) * (H_ * L_) + l;
      const float* hpp = &hp[b * H_];
      float ssum = 0.f;
#pragma unroll 4
      for (int h = 0; h < H_; ++h)
        ssum += V_l[h] * fast_tanh(hpp[h] + cp[(size_t)h * L_]);
      s_l[b * L_ + l] = ssum;
    }
    __syncthreads();

    // ---- masked softmax + pointer argmax (one wave per 2 rows) ----
#pragma unroll 1
    for (int rh = 0; rh < 2; ++rh) {
      int b = wave * 2 + rh;
      float vals[4], es[4];
      float mx = NEGBIG;
#pragma unroll
      for (int it = 0; it < 4; ++it) {
        int l = lane + 32 * it;
        float a = (mask_l[b * L_ + l] != 0.f) ? s_l[b * L_ + l] : NEGBIG;
        vals[it] = a;
        mx = fmaxf(mx, a);
      }
#pragma unroll
      for (int off = 16; off; off >>= 1) mx = fmaxf(mx, __shfl_xor(mx, off, 32));
      float sum = 0.f;
#pragma unroll
      for (int it = 0; it < 4; ++it) {
        float e = (vals[it] < -9.0e29f) ? 0.f : __expf(vals[it] - mx);
        es[it] = e;
        sum += e;
      }
#pragma unroll
      for (int off = 16; off; off >>= 1) sum += __shfl_xor(sum, off, 32);
      float inv = __builtin_amdgcn_rcpf(sum);
      float bv = -1.f;
      int bi_ = 0;
#pragma unroll
      for (int it = 0; it < 4; ++it) {
        int l = lane + 32 * it;
        float al = es[it] * inv;
        alpha_l[b * L_ + l] = al;
        outA[(size_t)(b0 + b) * (L_ * L_) + (size_t)t * L_ + l] = al;
        float mm = al * mask_l[b * L_ + l];
        if (mm > bv) { bv = mm; bi_ = l; }       // ascending l -> first max
      }
#pragma unroll
      for (int off = 16; off; off >>= 1) {
        float ov = __shfl_xor(bv, off, 32);
        int   oi = __shfl_xor(bi_, off, 32);
        if (ov > bv || (ov == bv && oi < bi_)) { bv = ov; bi_ = oi; }
      }
      if (lane == 0) {
        mask_l[b * L_ + bi_] = 0.f;
        outP[(size_t)(b0 + b) * L_ + t] = (float)bi_;
      }
    }
    __syncthreads();

    // ---- hs[b][h] = sum_l ctx[b][h][l] * alpha[b][l] ----
#pragma unroll 1
    for (int k = 0; k < 32; ++k) {
      int idx = tid + 256 * k;
      int b = idx >> 9, h = idx & 511;
      const float* cp = ctx + (size_t)(b0 + b) * (H_ * L_) + (size_t)h * L_;
      const float* ap = &alpha_l[b * L_];
      float acc = 0.f;
#pragma unroll 4
      for (int l = 0; l < L_; ++l) acc += cp[l] * ap[l];
      actB[b * H_ + h] = (bf16_t)acc;            // reuse actB as hs (bf16)
    }
    __syncthreads();

    // ---- GEMM3: h_new = tanh(hs @ (Wout_l+Wout_r).T + b_out) ----
#pragma unroll 1
    for (int tt = 0; tt < 4; ++tt) {
      int n0 = wave * 64 + tt * 16;
      f32x8 acc = wmma_row(actB, Wo, n0, lane);
      int n = n0 + (lane & 15);
      int mb = (lane >> 4) * 8;
      float bo = b_out[n];
#pragma unroll
      for (int r = 0; r < 8; ++r) {
        int m = mb + r;
        float hv = fast_tanh(acc[r] + bo);
        actA[m * H_ + n] = (bf16_t)hv;           // next-step h carry
        if (t == L_ - 1) outHf[(size_t)(b0 + m) * H_ + n] = hv;
      }
    }
    __syncthreads();
  }
}

// ---------------------------------------------------------------------------
// Host-side launch.  Workspace layout (bytes):
//   [0)            ctx   fp32  B*H*L        = 64 MiB
//   [67108864)     gx    fp32  B*4H         =  2 MiB
//   [69206016)     Whh   bf16  2048*512     =  2 MiB
//   [71303168)     Winp  bf16  512*512
//   [71827456)     Wctx  bf16  512*512
//   [72351744)     Wo    bf16  512*512      -> ~72.9 MiB total
// Output: [alphas B*L*L | pointers B*L (as float) | hf B*H | cf B*H]
// ---------------------------------------------------------------------------
extern "C" void kernel_launch(void* const* d_in, const int* in_sizes, int n_in,
                              void* d_out, int out_size, void* d_ws, size_t ws_size,
                              hipStream_t stream) {
  const float* dec_inp = (const float*)d_in[1];
  const float* h0      = (const float*)d_in[2];
  const float* c0      = (const float*)d_in[3];
  const float* context = (const float*)d_in[4];
  const float* W_ih    = (const float*)d_in[5];
  const float* b_ih    = (const float*)d_in[6];
  const float* W_hh    = (const float*)d_in[7];
  const float* b_hh    = (const float*)d_in[8];
  const float* W_out   = (const float*)d_in[9];
  const float* b_out   = (const float*)d_in[10];
  const float* W_inp   = (const float*)d_in[11];
  const float* b_inp   = (const float*)d_in[12];
  const float* W_ctx   = (const float*)d_in[13];
  const float* b_ctx   = (const float*)d_in[14];
  const float* V       = (const float*)d_in[15];

  char* ws = (char*)d_ws;
  float*  ctx  = (float*)(ws);
  float*  gx   = (float*)(ws + 67108864);
  bf16_t* Whh  = (bf16_t*)(ws + 69206016);
  bf16_t* Winp = (bf16_t*)(ws + 71303168);
  bf16_t* Wctx = (bf16_t*)(ws + 71827456);
  bf16_t* Wo   = (bf16_t*)(ws + 72351744);

  conv_kernel<<<7168, 256, 0, stream>>>(W_hh, W_inp, W_ctx, W_out,
                                        Whh, Winp, Wctx, Wo);
  gx_kernel<<<2048, 256, 0, stream>>>(dec_inp, W_ih, b_ih, b_hh, gx);
  ctx_kernel<<<2048, 256, 0, stream>>>(context, Wctx, b_ctx, ctx);

  float* out = (float*)d_out;
  float* outA  = out;                       // B*L*L = 4194304
  float* outP  = out + 4194304;             // B*L   = 32768
  float* outHf = out + 4227072;             // B*H   = 131072
  float* outCf = out + 4358144;             // B*H   = 131072
  scan_kernel<<<16, 256, 0, stream>>>(h0, c0, ctx, gx, Whh, Winp, Wo,
                                      b_inp, b_out, V,
                                      outA, outP, outHf, outCf);
}
